// LocalEmbedBlock_22093311770773
// MI455X (gfx1250) — compile-verified, matrix operand order checked by reading, not run on hardware
//
#include <hip/hip_runtime.h>
#include <hip/hip_bf16.h>
#include <math.h>

#define NPTS 1024
#define BATCH 8
#define CDIM 3
#define FDIM 128
#define HDIM 256
#define KNN 16
#define IN1 (2*FDIM)   // 256
#define OUT1 (2*HDIM)  // 512
#define IN2 OUT1       // 512
#define OUT2 HDIM      // 256
#define PPB 4          // points (M-tiles) per block: amortizes weight L2 traffic 4x

typedef __attribute__((ext_vector_type(16))) _Float16 v16h;
typedef __attribute__((ext_vector_type(8)))  _Float16 v8h;
typedef __attribute__((ext_vector_type(8)))  float    v8f;

union Frag16 { v16h v; v8h h[2]; };

__device__ __forceinline__ float gelu_exact(float x) {
    return 0.5f * x * (1.0f + erff(x * 0.70710678118654752440f));
}

// ---------------------------------------------------------------------------
// Kernel 1: fp32 -> f16 weight conversion (weights stay resident in L2)
// ---------------------------------------------------------------------------
__global__ void convert_weights_kernel(const float* __restrict__ W1,
                                       const float* __restrict__ W2,
                                       _Float16* __restrict__ W1h,
                                       _Float16* __restrict__ W2h) {
    int i = blockIdx.x * blockDim.x + threadIdx.x;
    if (i < OUT1 * IN1) W1h[i] = (_Float16)W1[i];   // 512*256
    if (i < OUT2 * IN2) W2h[i] = (_Float16)W2[i];   // 256*512
}

// ---------------------------------------------------------------------------
// Kernel 2: KNN (K=16 nearest, excluding self) per (b, n) row.
// One thread per row; the batch's 1024x3 points staged in LDS.
// Ascending-j insertion with strict '<' reproduces jax top_k tie order.
// ---------------------------------------------------------------------------
__global__ __launch_bounds__(256) void knn_kernel(const float* __restrict__ points,
                                                  int* __restrict__ idxbuf) {
    __shared__ float sp[NPTS * 3];
    const int tid = blockIdx.x * blockDim.x + threadIdx.x; // 0..8191
    const int b = tid >> 10;            // block-uniform (256 | 1024)
    const int n = tid & (NPTS - 1);

    for (int i = threadIdx.x; i < NPTS; i += blockDim.x) {
        sp[i * 3 + 0] = points[i * (BATCH * CDIM) + b * CDIM + 0];
        sp[i * 3 + 1] = points[i * (BATCH * CDIM) + b * CDIM + 1];
        sp[i * 3 + 2] = points[i * (BATCH * CDIM) + b * CDIM + 2];
    }
    __syncthreads();

    const float px = sp[n * 3 + 0], py = sp[n * 3 + 1], pz = sp[n * 3 + 2];

    float bd[KNN];
    int   bi[KNN];
#pragma unroll
    for (int k = 0; k < KNN; ++k) { bd[k] = 3.4e38f; bi[k] = -1; }

    for (int j = 0; j < NPTS; ++j) {
        if (j == n) continue;
        const float dx = sp[j * 3 + 0] - px;
        const float dy = sp[j * 3 + 1] - py;
        const float dz = sp[j * 3 + 2] - pz;
        const float d = dx * dx + dy * dy + dz * dz;
        if (d < bd[KNN - 1]) {
            int p = KNN - 1;
            while (p > 0 && bd[p - 1] > d) {
                bd[p] = bd[p - 1]; bi[p] = bi[p - 1]; --p;
            }
            bd[p] = d; bi[p] = j;
        }
    }
#pragma unroll
    for (int k = 0; k < KNN; ++k) idxbuf[tid * KNN + k] = bi[k];
}

// ---------------------------------------------------------------------------
// Kernel 3: gather + MLP via WMMA. One block (8 waves) per 4 (b, n) points.
// Each B (weight) fragment load from L2 feeds 4 WMMAs (one per point),
// cutting weight traffic 4x. 4 independent accumulation chains per B-tile
// also hide the WMMA->WMMA RAW hazard latency.
// ---------------------------------------------------------------------------
__global__ __launch_bounds__(256) void mlp_kernel(const float* __restrict__ features,
                                                  const float* __restrict__ b1,
                                                  const float* __restrict__ b2,
                                                  const _Float16* __restrict__ W1h,
                                                  const _Float16* __restrict__ W2h,
                                                  const int* __restrict__ idxbuf,
                                                  float* __restrict__ out) {
    __shared__ int      s_idx[PPB][KNN];
    __shared__ float    s_center[PPB][FDIM];
    __shared__ _Float16 Als[PPB][KNN * IN1];     // 4 x 16x256 f16 = 32 KB
    __shared__ _Float16 X1ls[PPB][KNN * OUT1];   // 4 x 16x512 f16 = 64 KB

    const int bn0 = blockIdx.x * PPB;       // first (b*N + n) of this block
    const int b  = bn0 >> 10;               // block-uniform (PPB | 1024)
    const int n0 = bn0 & (NPTS - 1);
    const int t = threadIdx.x;

    if (t < PPB * KNN)
        s_idx[t >> 4][t & 15] = idxbuf[(bn0 + (t >> 4)) * KNN + (t & 15)];
    if (t < FDIM) {
#pragma unroll
        for (int p = 0; p < PPB; ++p)
            s_center[p][t] = features[(n0 + p) * (BATCH * FDIM) + b * FDIM + t];
    }
    __syncthreads();

    // Build A tiles (row-major f16): cols 0..127 = nbr - center, 128..255 = center
    {
        const int m  = t >> 4;              // neighbor row 0..15
        const int c0 = (t & 15) * 16;       // 16 contiguous cols
#pragma unroll
        for (int p = 0; p < PPB; ++p) {
            const int id = s_idx[p][m];
#pragma unroll
            for (int c = 0; c < 16; ++c) {
                const int col = c0 + c;
                float val;
                if (col < FDIM) val = features[id * (BATCH * FDIM) + b * FDIM + col] - s_center[p][col];
                else            val = s_center[p][col - FDIM];
                Als[p][m * IN1 + col] = (_Float16)val;
            }
        }
    }
    __syncthreads();

    const int lane = t & 31;
    const int wv   = t >> 5;                // wave 0..7
    const int mrow = lane & 15;             // A: row M / B: col N / D: col N
    const int hi   = lane >> 4;             // selects K sub-block per 16-bit layout

    // -------- GEMM1: (4x)16x256 @ 256x512 -> (4x)16x512, wave owns 64 cols --
    v8f acc1[PPB][4] = {};
#pragma unroll
    for (int ks = 0; ks < IN1 / 32; ++ks) { // 8 k-steps
        const int k0 = ks * 32 + hi * 8;
        Frag16 a[PPB];
#pragma unroll
        for (int p = 0; p < PPB; ++p) {
            a[p].h[0] = *(const v8h*)&Als[p][mrow * IN1 + k0];
            a[p].h[1] = *(const v8h*)&Als[p][mrow * IN1 + k0 + 16];
        }
#pragma unroll
        for (int tt = 0; tt < 4; ++tt) {
            const int colB = wv * 64 + tt * 16 + mrow;  // output column (row of W1)
            Frag16 bm;
            bm.h[0] = *(const v8h*)&W1h[colB * IN1 + k0];
            bm.h[1] = *(const v8h*)&W1h[colB * IN1 + k0 + 16];
#pragma unroll
            for (int p = 0; p < PPB; ++p)
                acc1[p][tt] = __builtin_amdgcn_wmma_f32_16x16x32_f16(
                    false, a[p].v, false, bm.v, (short)0, acc1[p][tt], false, false);
        }
    }
    // bias + GELU -> X1 (f16, row-major in LDS)
#pragma unroll
    for (int tt = 0; tt < 4; ++tt) {
        const int col = wv * 64 + tt * 16 + mrow;
        const float bias = b1[col];
#pragma unroll
        for (int p = 0; p < PPB; ++p) {
#pragma unroll
            for (int r = 0; r < 8; ++r) {
                const float g = gelu_exact(acc1[p][tt][r] + bias);
                const int m2 = r + hi * 8;  // C/D layout: VGPR r -> row, lane -> col
                X1ls[p][m2 * OUT1 + col] = (_Float16)g;
            }
        }
    }
    __syncthreads();

    // -------- GEMM2: (4x)16x512 @ 512x256 -> (4x)16x256, wave owns 32 cols --
    v8f acc2[PPB][2] = {};
#pragma unroll
    for (int ks = 0; ks < IN2 / 32; ++ks) { // 16 k-steps
        const int k0 = ks * 32 + hi * 8;
        Frag16 a[PPB];
#pragma unroll
        for (int p = 0; p < PPB; ++p) {
            a[p].h[0] = *(const v8h*)&X1ls[p][mrow * IN2 + k0];
            a[p].h[1] = *(const v8h*)&X1ls[p][mrow * IN2 + k0 + 16];
        }
#pragma unroll
        for (int tt = 0; tt < 2; ++tt) {
            const int colB = wv * 32 + tt * 16 + mrow;
            Frag16 bm;
            bm.h[0] = *(const v8h*)&W2h[colB * IN2 + k0];
            bm.h[1] = *(const v8h*)&W2h[colB * IN2 + k0 + 16];
#pragma unroll
            for (int p = 0; p < PPB; ++p)
                acc2[p][tt] = __builtin_amdgcn_wmma_f32_16x16x32_f16(
                    false, a[p].v, false, bm.v, (short)0, acc2[p][tt], false, false);
        }
    }
    // bias + GELU + mean over 16 neighbor rows, write (N, B, H) output
#pragma unroll
    for (int tt = 0; tt < 2; ++tt) {
        const int col = wv * 32 + tt * 16 + mrow;
        const float bias = b2[col];
#pragma unroll
        for (int p = 0; p < PPB; ++p) {
            float partial = 0.0f;
#pragma unroll
            for (int r = 0; r < 8; ++r)
                partial += gelu_exact(acc2[p][tt][r] + bias);
            partial += __shfl_xor(partial, 16, 32); // combine rows 0-7 with 8-15
            if (hi == 0)
                out[(n0 + p) * (BATCH * HDIM) + b * HDIM + col] = partial * (1.0f / 16.0f);
        }
    }
}

// ---------------------------------------------------------------------------
extern "C" void kernel_launch(void* const* d_in, const int* in_sizes, int n_in,
                              void* d_out, int out_size, void* d_ws, size_t ws_size,
                              hipStream_t stream) {
    const float* points   = (const float*)d_in[0]; // (N,B,C)
    const float* features = (const float*)d_in[1]; // (N,B,F)
    const float* W1       = (const float*)d_in[2]; // (512,256)
    const float* b1       = (const float*)d_in[3]; // (512)
    const float* W2       = (const float*)d_in[4]; // (256,512)
    const float* b2       = (const float*)d_in[5]; // (256)
    float* out = (float*)d_out;                    // (N,B,H)

    char* ws = (char*)d_ws;
    int*      idxbuf = (int*)ws;                                  // 8*1024*16*4 = 512 KB
    _Float16* W1h    = (_Float16*)(ws + 512 * 1024);              // 256 KB
    _Float16* W2h    = (_Float16*)(ws + 512 * 1024 + 256 * 1024); // 256 KB

    convert_weights_kernel<<<(OUT1 * IN1 + 255) / 256, 256, 0, stream>>>(W1, W2, W1h, W2h);
    knn_kernel<<<(BATCH * NPTS) / 256, 256, 0, stream>>>(points, idxbuf);
    mlp_kernel<<<(BATCH * NPTS) / PPB, 256, 0, stream>>>(features, b1, b2, W1h, W2h, idxbuf, out);
}
